// PFNN_54846732370279
// MI455X (gfx1250) — matrix operand order repaired and checked by reading, not compiled, not run on hardware
//
#include <hip/hip_runtime.h>
#include <math.h>

// ---------------------------------------------------------------------------
// PFNN on MI455X (gfx1250). Catmull-Rom blend coeffs folded into the A matrix
// so each layer is ONE dense bf16 GEMM (v_wmma_f32_16x16x32_bf16, f32 accum).
// GEMM uses double-buffered LDS tiles filled by gfx1250 async-to-LDS copies
// (global_load_async_to_lds_b128 + s_wait_asynccnt) to hide L2 latency at the
// low occupancy this tiny problem provides. Layer GEMMs fuse bias-blend + ELU
// + construction of the NEXT layer's folded A directly in the epilogue.
// ---------------------------------------------------------------------------

typedef __attribute__((ext_vector_type(16))) __bf16 v16bf;
typedef __attribute__((ext_vector_type(8)))  __bf16 v8bf;
typedef __attribute__((ext_vector_type(8)))  float  v8f;
typedef __attribute__((ext_vector_type(4)))  float  v4f;

#define BATCH    256
#define IN_DIM   342
#define HID      512
#define OUT_DIM  311
#define NSLICES  4
#define KP0      352            // IN_DIM padded to multiple of 32
#define KT0      (NSLICES*KP0)  // 1408
#define KT12     (NSLICES*HID)  // 2048
#define NP2      384            // OUT_DIM padded to multiple of 128

#define LDSOFF(p) ((unsigned)(unsigned long long)(p))

__device__ __forceinline__ void async_cp16(unsigned lds_addr, const void* gaddr) {
    // gfx1250 async copy: 16B global -> LDS, tracked by ASYNCcnt
    asm volatile("global_load_async_to_lds_b128 %0, %1, off"
                 :: "v"(lds_addr), "v"(gaddr) : "memory");
}
__device__ __forceinline__ void wait_async_le3() {
    asm volatile("s_wait_asynccnt 0x3" ::: "memory");
}
__device__ __forceinline__ void wait_async_0() {
    asm volatile("s_wait_asynccnt 0x0" ::: "memory");
}

// --- Weight transpose+convert: Wt[n][s*Kp+k] = bf16(w[s][k][n]) -------------
// 3-D grid (no div/mod): x covers k, y = n (padded), z = s.
__global__ void pfnn_wt_k(const float* __restrict__ w, __bf16* __restrict__ wt,
                          int K, int N, int Kp) {
    int k = blockIdx.x * blockDim.x + threadIdx.x;
    int n = blockIdx.y;
    int s = blockIdx.z;
    if (k >= Kp) return;
    float v = (n < N && k < K) ? w[((long)s * K + k) * N + n] : 0.0f;
    wt[(long)n * (NSLICES * (long)Kp) + (long)s * Kp + k] = (__bf16)v;
}

// --- Fused: Catmull-Rom coeffs + folded A0 (one block per sample) -----------
__global__ __launch_bounds__(256)
void pfnn_a0_k(const float* __restrict__ x, float* __restrict__ coeff,
               __bf16* __restrict__ A0) {
    int b = blockIdx.x;
    __shared__ float c[4];
    if (threadIdx.x == 0) {
        float phase = x[b * (IN_DIM + 1) + IN_DIM];
        float ps  = (float)NSLICES * phase;
        float mu  = ps - floorf(ps);
        int   p1  = ((int)ps) & 3;
        int   p0  = (p1 + 3) & 3, p2 = (p1 + 1) & 3, p3 = (p1 + 2) & 3;
        float mu2 = mu * mu, mu3 = mu2 * mu;
        c[p0] = -0.5f * mu3 + mu2 - 0.5f * mu;
        c[p1] =  1.5f * mu3 - 2.5f * mu2 + 1.0f;
        c[p2] = -1.5f * mu3 + 2.0f * mu2 + 0.5f * mu;
        c[p3] =  0.5f * mu3 - 0.5f * mu2;
        float* cg = coeff + b * 4;
        cg[0] = c[0]; cg[1] = c[1]; cg[2] = c[2]; cg[3] = c[3];
    }
    __syncthreads();
    const float* xr = x + (long)b * (IN_DIM + 1);
    __bf16* ar = A0 + (long)b * KT0;
#pragma unroll
    for (int s = 0; s < NSLICES; ++s) {
        float cs = c[s];
        for (int k = threadIdx.x; k < KP0; k += 256) {
            float v = (k < IN_DIM) ? cs * xr[k] : 0.0f;
            ar[s * KP0 + k] = (__bf16)v;
        }
    }
}

// --- bf16 WMMA GEMM with async double-buffered LDS staging ------------------
// Block = 8 waves, tile 64(M) x 128(N); wave tile 32x32 (2x2 WMMA accums).
// FUSE: epilogue writes next layer's folded A (c[m,s]*elu(y)) instead of Y.
template<bool FUSE, bool DOELU>
__global__ __launch_bounds__(256)
void pfnn_gemm_k(const __bf16* __restrict__ A, const __bf16* __restrict__ Wt,
                 int Kt, const float* __restrict__ coeff,
                 const float* __restrict__ bias, int N,
                 float* __restrict__ Y, int ldy, __bf16* __restrict__ Anext) {
    __shared__ __align__(16) __bf16 shA[2][64][40];    // 80B rows (bank-spread)
    __shared__ __align__(16) __bf16 shB[2][128][40];

    const int t    = threadIdx.x;
    const int lane = t & 31;
    const int wave = t >> 5;
    const int half = lane >> 4;
    const int ln   = lane & 15;
    const int blockM = blockIdx.y * 64;
    const int blockN = blockIdx.x * 128;
    const int m0 = blockM + (wave & 1) * 32;
    const int n0 = blockN + (wave >> 1) * 32;

    // Copy assignment: 256 threads cover A tile (64 rows x 4 chunks of 16B)
    // and B tile (128 rows x 4 chunks, 2 rows per thread).
    const int crow = t >> 2;            // 0..63
    const int cchk = (t & 3) * 8;       // element offset 0,8,16,24
    const __bf16* gA  = A  + (long)(blockM + crow) * Kt + cchk;
    const __bf16* gB0 = Wt + (long)(blockN + crow) * Kt + cchk;
    const __bf16* gB1 = Wt + (long)(blockN + crow + 64) * Kt + cchk;

    v8f acc[2][2] = {};

    // Prologue: stage k0 = 0 into buffer 0.
    async_cp16(LDSOFF(&shA[0][crow][cchk]),      gA);
    async_cp16(LDSOFF(&shB[0][crow][cchk]),      gB0);
    async_cp16(LDSOFF(&shB[0][crow + 64][cchk]), gB1);

    for (int k0 = 0; k0 < Kt; k0 += 32) {
        const int buf = (k0 >> 5) & 1;
        if (k0 + 32 < Kt) {
            const int nb = buf ^ 1;
            async_cp16(LDSOFF(&shA[nb][crow][cchk]),      gA  + k0 + 32);
            async_cp16(LDSOFF(&shB[nb][crow][cchk]),      gB0 + k0 + 32);
            async_cp16(LDSOFF(&shB[nb][crow + 64][cchk]), gB1 + k0 + 32);
            wait_async_le3();        // current buffer's 3 copies complete
        } else {
            wait_async_0();
        }
        __syncthreads();

        // A fragment (16x32): lane needs K in [8h,8h+8) u [16+8h,24+8h)
        v16bf afr[2], bfr[2];
#pragma unroll
        for (int mi = 0; mi < 2; ++mi) {
            const __bf16* ar = &shA[buf][(wave & 1) * 32 + mi * 16 + ln][0];
            v8bf lo = *(const v8bf*)(ar + 8 * half);
            v8bf hi = *(const v8bf*)(ar + 16 + 8 * half);
            afr[mi] = __builtin_shufflevector(lo, hi,
                      0, 1, 2, 3, 4, 5, 6, 7, 8, 9, 10, 11, 12, 13, 14, 15);
        }
        // B fragment (32x16): lane = column, K in [8h,8h+8) u [16+8h,24+8h)
#pragma unroll
        for (int ni = 0; ni < 2; ++ni) {
            const __bf16* br = &shB[buf][(wave >> 1) * 32 + ni * 16 + ln][0];
            v8bf lo = *(const v8bf*)(br + 8 * half);
            v8bf hi = *(const v8bf*)(br + 16 + 8 * half);
            bfr[ni] = __builtin_shufflevector(lo, hi,
                      0, 1, 2, 3, 4, 5, 6, 7, 8, 9, 10, 11, 12, 13, 14, 15);
        }
#pragma unroll
        for (int mi = 0; mi < 2; ++mi)
#pragma unroll
            for (int ni = 0; ni < 2; ++ni)
                acc[mi][ni] = __builtin_amdgcn_wmma_f32_16x16x32_bf16(
                    false, afr[mi], false, bfr[ni], (short)0, acc[mi][ni],
                    false, false);
        __syncthreads();             // reads done before buffer reuse
    }

    // Epilogue: bias blend (unconditional clamped loads), ELU, store.
    float bs[2][4];
#pragma unroll
    for (int ni = 0; ni < 2; ++ni) {
        int n  = n0 + ni * 16 + ln;
        int nc = n < N ? n : N - 1;             // clamp: loads never branch
#pragma unroll
        for (int s = 0; s < 4; ++s) bs[ni][s] = bias[(long)s * N + nc];
    }
#pragma unroll
    for (int mi = 0; mi < 2; ++mi) {
#pragma unroll
        for (int v = 0; v < 8; ++v) {
            int m = m0 + mi * 16 + 8 * half + v;   // D: VGPR v -> row v+8*half
            v4f cf = *(const v4f*)(coeff + m * 4);
#pragma unroll
            for (int ni = 0; ni < 2; ++ni) {
                int n = n0 + ni * 16 + ln;
                float y = acc[mi][ni][v]
                        + cf[0] * bs[ni][0] + cf[1] * bs[ni][1]
                        + cf[2] * bs[ni][2] + cf[3] * bs[ni][3];
                if (DOELU) y = (y > 0.0f) ? y : expm1f(y);
                if (FUSE) {
                    // write next layer's folded A: A'[m][s*HID+n] = c[s]*y
                    __bf16* dst = Anext + (long)m * KT12 + n;
#pragma unroll
                    for (int s = 0; s < 4; ++s)
                        dst[(long)s * HID] = (__bf16)(cf[s] * y);
                } else {
                    if (n < N) Y[(long)m * ldy + n] = y;
                }
            }
        }
    }
}

extern "C" void kernel_launch(void* const* d_in, const int* in_sizes, int n_in,
                              void* d_out, int out_size, void* d_ws, size_t ws_size,
                              hipStream_t stream) {
    const float* x  = (const float*)d_in[0];
    const float* w0 = (const float*)d_in[1];
    const float* b0 = (const float*)d_in[2];
    const float* w1 = (const float*)d_in[3];
    const float* b1 = (const float*)d_in[4];
    const float* w2 = (const float*)d_in[5];
    const float* b2 = (const float*)d_in[6];
    float* out = (float*)d_out;
    (void)in_sizes; (void)n_in; (void)out_size; (void)ws_size;

    char* ws = (char*)d_ws;
    size_t off = 0;
    auto take = [&](size_t bytes) {
        char* p = ws + off;
        off = (off + bytes + 255) & ~(size_t)255;
        return p;
    };
    float*  coeff = (float*) take((size_t)BATCH * 4 * sizeof(float));
    __bf16* A0    = (__bf16*)take((size_t)BATCH * KT0  * 2);
    __bf16* Wt0   = (__bf16*)take((size_t)HID   * KT0  * 2);
    __bf16* A1    = (__bf16*)take((size_t)BATCH * KT12 * 2);
    __bf16* Wt1   = (__bf16*)take((size_t)HID   * KT12 * 2);
    __bf16* A2    = (__bf16*)take((size_t)BATCH * KT12 * 2);
    __bf16* Wt2   = (__bf16*)take((size_t)NP2   * KT12 * 2);

    // Weight transpose/convert (3-D grids, no integer division).
    pfnn_wt_k<<<dim3((KP0 + 255) / 256, HID, NSLICES), 256, 0, stream>>>(
        w0, Wt0, IN_DIM, HID, KP0);
    pfnn_wt_k<<<dim3((HID + 255) / 256, HID, NSLICES), 256, 0, stream>>>(
        w1, Wt1, HID, HID, HID);
    pfnn_wt_k<<<dim3((HID + 255) / 256, NP2, NSLICES), 256, 0, stream>>>(
        w2, Wt2, HID, OUT_DIM, HID);

    // Coeffs + folded A0.
    pfnn_a0_k<<<BATCH, 256, 0, stream>>>(x, coeff, A0);

    dim3 g01(HID / 128, BATCH / 64);   // 4 x 4
    dim3 g2 (NP2 / 128, BATCH / 64);   // 3 x 4

    // Layer 0: GEMM + bias + ELU, fused emit of folded A1.
    pfnn_gemm_k<true, true><<<g01, 256, 0, stream>>>(
        A0, Wt0, KT0, coeff, b0, HID, nullptr, 0, A1);
    // Layer 1: GEMM + bias + ELU, fused emit of folded A2.
    pfnn_gemm_k<true, true><<<g01, 256, 0, stream>>>(
        A1, Wt1, KT12, coeff, b1, HID, nullptr, 0, A2);
    // Layer 2: GEMM + bias, write final output (N = 311 guarded).
    pfnn_gemm_k<false, false><<<g2, 256, 0, stream>>>(
        A2, Wt2, KT12, coeff, b2, OUT_DIM, out, OUT_DIM, nullptr);
}
